// SoftSAE_37830071943565
// MI455X (gfx1250) — matrix operand order repaired and checked by reading, not compiled, not run on hardware
//
#include <hip/hip_runtime.h>
#include <hip/hip_bf16.h>
#include <stdint.h>

typedef __attribute__((ext_vector_type(16))) __bf16 v16bf;
typedef __attribute__((ext_vector_type(8)))  float  v8f;

#define WAVE   32
#define BM     128
#define BN     64
#define BK     32
#define NSUB   4      // BN / 16 subtiles per wave

union V16 { v16bf v; uint4 q[2]; };

#if __has_builtin(__builtin_amdgcn_global_load_async_to_lds_b128)
#define HAVE_ASYNC_LDS 1
typedef int v4i_vs __attribute__((vector_size(16)));
typedef __attribute__((address_space(1))) v4i_vs as1_v4i;
typedef __attribute__((address_space(3))) v4i_vs as3_v4i;
#endif

__device__ __forceinline__ void wait_async_then_barrier()
{
#ifdef HAVE_ASYNC_LDS
#if __has_builtin(__builtin_amdgcn_s_wait_asynccnt)
    __builtin_amdgcn_s_wait_asynccnt(0);
#else
    asm volatile("s_wait_asynccnt 0" ::: "memory");
#endif
#endif
    __syncthreads();
}

// ---------------------------------------------------------------------------
// Tiled bf16 GEMM  C[M,N] = A[M,K] @ W[N,K]^T  using v_wmma_f32_16x16x32_bf16.
// Tiles staged global->LDS with GLOBAL_LOAD_ASYNC_TO_LDS_B128 (ASYNCcnt path)
// when available; falls back to load+ds_store otherwise.
// EPI 0: out = bf16( relu(C + bias) )            (encoder pre-activations)
// EPI 1: per-row partial of relu(C+bias) . w2    (k-estimator fused reduce)
// EPI 2: out = f32( C + bias )                   (decoder)
// 256 threads = 8 waves; each wave owns a 16-row strip and NSUB 16x16 tiles.
// ---------------------------------------------------------------------------
template<int EPI>
__global__ __launch_bounds__(256)
void gemm_bf16(const __bf16* __restrict__ A, const __bf16* __restrict__ W,
               const float* __restrict__ bias, const float* __restrict__ w2,
               __bf16* __restrict__ outBf, float* __restrict__ outF,
               float* __restrict__ partial,
               int M, int N, int K, int npart)
{
    __shared__ __bf16 sA[BM][BK];   // 8 KB
    __shared__ __bf16 sB[BN][BK];   // 4 KB

    const int t   = threadIdx.x;
    const int w   = t / WAVE;       // wave id 0..7
    const int L   = t % WAVE;       // lane
    const int l16 = L % 16;
    const int hi  = L / 16;
    const int m0  = blockIdx.x * BM;
    const int n0  = blockIdx.y * BN;

    v8f acc[NSUB] = {};

    for (int k0 = 0; k0 < K; k0 += BK) {
#ifdef HAVE_ASYNC_LDS
        // --- async DMA global -> LDS (no VGPR round-trip, ASYNCcnt) ---
        #pragma unroll
        for (int i = 0; i < 2; ++i) {
            int idx = t + i * 256;
            int row = idx >> 2;             // 4 x b128 per 32-elem row
            int col = (idx & 3) * 8;
            __builtin_amdgcn_global_load_async_to_lds_b128(
                (as1_v4i*)(A + (size_t)(m0 + row) * K + k0 + col),
                (as3_v4i*)&sA[row][col], 0, 0);
        }
        {
            int row = t >> 2;
            int col = (t & 3) * 8;
            __builtin_amdgcn_global_load_async_to_lds_b128(
                (as1_v4i*)(W + (size_t)(n0 + row) * K + k0 + col),
                (as3_v4i*)&sB[row][col], 0, 0);
        }
#else
        // --- stage A tile: 128x32 bf16 = 512 uint4, 2 per thread ---
        #pragma unroll
        for (int i = 0; i < 2; ++i) {
            int idx = t + i * 256;
            int row = idx >> 2;
            int col = (idx & 3) * 8;
            *reinterpret_cast<uint4*>(&sA[row][col]) =
                *reinterpret_cast<const uint4*>(A + (size_t)(m0 + row) * K + k0 + col);
        }
        // --- stage B tile: 64x32 bf16 = 256 uint4, 1 per thread ---
        {
            int row = t >> 2;
            int col = (t & 3) * 8;
            *reinterpret_cast<uint4*>(&sB[row][col]) =
                *reinterpret_cast<const uint4*>(W + (size_t)(n0 + row) * K + k0 + col);
        }
#endif
        wait_async_then_barrier();

        // prefetch next K-tile into near caches (global_prefetch_b8, WGP scope)
        if (k0 + BK < K) {
            __builtin_prefetch(A + (size_t)(m0 + (t >> 1)) * K + k0 + BK, 0, 3);
            __builtin_prefetch(W + (size_t)(n0 + (t >> 2)) * K + k0 + BK, 0, 3);
        }

        // A fragment (16x32 bf16): lane<16 holds K {0..7,16..23}, lane>=16 {8..15,24..31}
        V16 ua;
        int arow = 16 * w + l16;
        ua.q[0] = *reinterpret_cast<const uint4*>(&sA[arow][8 * hi]);
        ua.q[1] = *reinterpret_cast<const uint4*>(&sA[arow][16 + 8 * hi]);

        #pragma unroll
        for (int s = 0; s < NSUB; ++s) {
            // B fragment (32x16): lanes 0-15 K={0..7,16..23} of col n, lanes 16-31 K={8..15,24..31}
            V16 ub;
            int bcol = 16 * s + l16;
            ub.q[0] = *reinterpret_cast<const uint4*>(&sB[bcol][8 * hi]);
            ub.q[1] = *reinterpret_cast<const uint4*>(&sB[bcol][16 + 8 * hi]);
            acc[s] = __builtin_amdgcn_wmma_f32_16x16x32_bf16(
                         false, ua.v, false, ub.v, (short)0, acc[s], false, false);
        }
        __syncthreads();
    }

    // --- epilogue ---
    if (EPI == 1) {
        float rowsum[8];
        #pragma unroll
        for (int r = 0; r < 8; ++r) rowsum[r] = 0.f;
        #pragma unroll
        for (int s = 0; s < NSUB; ++s) {
            int ng = n0 + 16 * s + l16;
            float wv = w2[ng];
            float bv = bias[ng];
            #pragma unroll
            for (int r = 0; r < 8; ++r)
                rowsum[r] += fmaxf(acc[s][r] + bv, 0.f) * wv;
        }
        #pragma unroll
        for (int r = 0; r < 8; ++r) {
            float v = rowsum[r];
            #pragma unroll
            for (int msk = 1; msk < 16; msk <<= 1)
                v += __shfl_xor(v, msk, WAVE);
            if (l16 == 0) {
                int mg = m0 + 16 * w + r + 8 * hi;
                partial[(size_t)mg * npart + blockIdx.y] = v;  // deterministic
            }
        }
    } else {
        #pragma unroll
        for (int s = 0; s < NSUB; ++s) {
            int ng = n0 + 16 * s + l16;
            float bv = bias[ng];
            #pragma unroll
            for (int r = 0; r < 8; ++r) {
                int mg = m0 + 16 * w + r + 8 * hi;
                float v = acc[s][r] + bv;
                if (EPI == 0) {
                    outBf[(size_t)mg * N + ng] = (__bf16)fmaxf(v, 0.f);
                } else {
                    outF[(size_t)mg * N + ng] = v;
                }
            }
        }
    }
}

// ---------------------------------------------------------------------------
__global__ void cvt_f32_bf16(const float* __restrict__ src, __bf16* __restrict__ dst,
                             size_t n)
{
    size_t i = (size_t)blockIdx.x * blockDim.x + threadIdx.x;
    size_t stride = (size_t)gridDim.x * blockDim.x;
    for (; i < n; i += stride) dst[i] = (__bf16)src[i];
}

__global__ void make_xc(const float* __restrict__ x, const float* __restrict__ b_dec,
                        const float* __restrict__ nf, __bf16* __restrict__ xc,
                        int D, size_t n)
{
    float inv = 1.0f / nf[0];
    size_t i = (size_t)blockIdx.x * blockDim.x + threadIdx.x;
    size_t stride = (size_t)gridDim.x * blockDim.x;
    for (; i < n; i += stride)
        xc[i] = (__bf16)(x[i] * inv - b_dec[i % D]);
}

// logit[b] = sum(partial[b,:]); k[b] = clamp(int(exp(logit+b2)), 1, F)
__global__ __launch_bounds__(256)
void reduce_logit(const float* __restrict__ partial, const float* __restrict__ kb2,
                  int* __restrict__ kv, int npart, int F)
{
    int b = blockIdx.x;
    __shared__ float s[256];
    float v = 0.f;
    for (int j = threadIdx.x; j < npart; j += 256)
        v += partial[(size_t)b * npart + j];
    s[threadIdx.x] = v;
    __syncthreads();
    for (int st = 128; st > 0; st >>= 1) {
        if (threadIdx.x < st) s[threadIdx.x] += s[threadIdx.x + st];
        __syncthreads();
    }
    if (threadIdx.x == 0) {
        float kh = __expf(s[0] + kb2[0]);
        kh = fminf(fmaxf(kh, 1.0f), (float)F);
        kv[b] = (int)kh;
    }
}

// Per-row variable-k top-k via 2-pass 256-bin radix select on bf16 bit pattern
// (monotonic for non-negative values). Deterministic tie handling by index.
__global__ __launch_bounds__(256)
void topk_select(const unsigned short* __restrict__ pre, const int* __restrict__ kv,
                 unsigned short* __restrict__ enc, int F)
{
    int b = blockIdx.x;
    const unsigned short* row = pre + (size_t)b * F;
    unsigned short* out = enc + (size_t)b * F;
    int k = kv[b];

    __shared__ unsigned hist[256];
    __shared__ int cnt[256];
    __shared__ int s_hb, s_rem, s_T, s_rem2;
    int t = threadIdx.x;

    // pass 1: coarse bins on bits [15:8]
    hist[t] = 0;
    __syncthreads();
    for (int j = t; j < F; j += 256) atomicAdd(&hist[row[j] >> 8], 1u);
    __syncthreads();
    if (t == 0) {
        int c = 0, hb = 0, above = 0;
        for (int bn = 255; bn >= 0; --bn) {
            int nc = c + (int)hist[bn];
            if (nc >= k) { hb = bn; above = c; break; }
            c = nc;
        }
        s_hb = hb; s_rem = k - above;
    }
    __syncthreads();
    int hb = s_hb, rem = s_rem;

    // pass 2: fine bins on bits [7:0] within coarse bin
    hist[t] = 0;
    __syncthreads();
    for (int j = t; j < F; j += 256) {
        unsigned key = row[j];
        if ((int)(key >> 8) == hb) atomicAdd(&hist[key & 255u], 1u);
    }
    __syncthreads();
    if (t == 0) {
        int c = 0, lb = 0, above = 0;
        for (int bn = 255; bn >= 0; --bn) {
            int nc = c + (int)hist[bn];
            if (nc >= rem) { lb = bn; above = c; break; }
            c = nc;
        }
        s_T = (hb << 8) | lb; s_rem2 = rem - above;
    }
    __syncthreads();
    unsigned T = (unsigned)s_T;

    // deterministic rank of threshold ties: prefix over contiguous chunks
    int chunk = F / 256;                  // 128
    int beg = t * chunk, end = beg + chunk;
    int eq = 0;
    for (int j = beg; j < end; ++j) if ((unsigned)row[j] == T) ++eq;
    cnt[t] = eq;
    __syncthreads();
    if (t == 0) {
        int run = 0;
        for (int i = 0; i < 256; ++i) { int c = cnt[i]; cnt[i] = run; run += c; }
    }
    __syncthreads();
    int rank = cnt[t];
    int rem2 = s_rem2;
    for (int j = beg; j < end; ++j) {
        unsigned key = row[j];
        bool keep;
        if (key > T)       keep = true;
        else if (key == T) { keep = (rank < rem2); ++rank; }
        else               keep = false;
        out[j] = keep ? (unsigned short)key : (unsigned short)0;
    }
}

// ---------------------------------------------------------------------------
extern "C" void kernel_launch(void* const* d_in, const int* in_sizes, int n_in,
                              void* d_out, int out_size, void* d_ws, size_t ws_size,
                              hipStream_t stream)
{
    const int B = 4096, D = 768, F = 32768;
    const int NPART = F / BN;             // 512 N-blocks in kest GEMM

    const float* x     = (const float*)d_in[0];
    const float* enc_W = (const float*)d_in[1];
    const float* enc_b = (const float*)d_in[2];
    const float* dec_W = (const float*)d_in[3];
    const float* b_dec = (const float*)d_in[4];
    const float* kW1   = (const float*)d_in[5];
    const float* kb1   = (const float*)d_in[6];
    const float* kW2   = (const float*)d_in[7];
    const float* kb2   = (const float*)d_in[8];
    const float* nf    = (const float*)d_in[9];
    float* x_hat = (float*)d_out;

    char* ws = (char*)d_ws;
    size_t off = 0;
    auto take = [&](size_t bytes) -> void* {
        void* p = ws + off;
        off = (off + bytes + 255) & ~(size_t)255;
        return p;
    };
    __bf16* encWb  = (__bf16*)take((size_t)F * D * 2);   // 50 MB (L2-resident)
    __bf16* kW1b   = (__bf16*)take((size_t)F * D * 2);   // 50 MB
    __bf16* decWb  = (__bf16*)take((size_t)D * F * 2);   // 50 MB
    __bf16* xcb    = (__bf16*)take((size_t)B * D * 2);   // 6.3 MB
    float*  part   = (float*) take((size_t)B * NPART * 4); // 8.4 MB
    int*    kvals  = (int*)   take((size_t)B * 4);
    __bf16* preb   = (__bf16*)take((size_t)B * F * 2);   // 256 MB
    __bf16* encb   = (__bf16*)take((size_t)B * F * 2);   // 256 MB

    // 1) precision conversion (weights fit L2 afterwards) + centered input
    size_t nW = (size_t)F * D;
    cvt_f32_bf16<<<4096, 256, 0, stream>>>(enc_W, encWb, nW);
    cvt_f32_bf16<<<4096, 256, 0, stream>>>(kW1,   kW1b,  nW);
    cvt_f32_bf16<<<4096, 256, 0, stream>>>(dec_W, decWb, nW);
    make_xc<<<2048, 256, 0, stream>>>(x, b_dec, nf, xcb, D, (size_t)B * D);

    // 2) k-estimator GEMM with fused .W2 reduction into per-block partials
    gemm_bf16<1><<<dim3(B / BM, F / BN), 256, 0, stream>>>(
        xcb, kW1b, kb1, kW2, nullptr, nullptr, part, B, F, D, NPART);

    // 3) logit -> k per row (deterministic tree reduce)
    reduce_logit<<<B, 256, 0, stream>>>(part, kb2, kvals, NPART, F);

    // 4) encoder GEMM: pre = relu(xc @ enc_W.T + enc_b), stored bf16
    gemm_bf16<0><<<dim3(B / BM, F / BN), 256, 0, stream>>>(
        xcb, encWb, enc_b, nullptr, preb, nullptr, nullptr, B, F, D, 0);

    // 5) per-row variable-k top-k (radix select on bf16 bits)
    topk_select<<<B, 256, 0, stream>>>(
        (const unsigned short*)preb, kvals, (unsigned short*)encb, F);

    // 6) decoder GEMM: x_hat = enc @ dec_W.T + b_dec  (K = 32768)
    gemm_bf16<2><<<dim3(B / BM, D / BN), 256, 0, stream>>>(
        encb, decWb, b_dec, nullptr, nullptr, x_hat, nullptr, B, D, F, 0);
}
